// NHConv_274877907665
// MI455X (gfx1250) — matrix operand order repaired and checked by reading, not compiled, not run on hardware
//
#include <hip/hip_runtime.h>

#define BB 2
#define NN 196608
#define KK 9
#define FIN 64
#define FOUT 64

typedef __attribute__((ext_vector_type(16))) __bf16 v16bf;
typedef __attribute__((ext_vector_type(8)))  float  v8f;

// Kernel 1: convert W[k][f][o] (f32) -> Wt[k][o][f] (bf16) in workspace.
// 72KB total: stays resident in WGP$/L2 for the whole main kernel.
__global__ __launch_bounds__(256) void nhconv_prep_w(const float* __restrict__ W,
                                                     __bf16* __restrict__ Wt) {
  int idx = blockIdx.x * blockDim.x + threadIdx.x;
  if (idx >= KK * FIN * FOUT) return;
  int k = idx >> 12;          // / 4096
  int f = (idx >> 6) & 63;
  int o = idx & 63;
  Wt[(k << 12) + (o << 6) + f] = (__bf16)W[idx];
}

// Kernel 2: one wave computes a 32x64 output tile (two 16-row A tiles sharing
// the same B fragments -> 4 WMMAs per B-pair) via v_wmma_f32_16x16x32_bf16.
__global__ __launch_bounds__(256) void nhconv_main(
    const float*     __restrict__ x,      // [B][N][64] f32
    const long long* __restrict__ adjc,   // [N][9] i64
    const __bf16*    __restrict__ Wt,     // [9][64][64] bf16, [k][o][f]
    const float*     __restrict__ bias,   // [64]
    float*           __restrict__ out)    // [B][N][64] f32
{
  const int lane  = threadIdx.x & 31;
  const int wave  = threadIdx.x >> 5;
  const int gwave = blockIdx.x * (blockDim.x >> 5) + wave;
  const int nwav  = gridDim.x * (blockDim.x >> 5);

  const int col   = lane & 15;   // output column within a 16-wide tile
  const int hi    = lane >> 4;   // lane half
  const int abase = hi * 8;      // prefetch offset only (A fragments load at kb)
  const int kb    = hi * 16;     // B-fragment K base    (contiguous k = 16*hi + s; A uses it too)

  float bcol[4];
#pragma unroll
  for (int j = 0; j < 4; ++j) bcol[j] = bias[j * 16 + col];

  const int tilesTotal = BB * (NN / 32);   // 12288 tiles of 32 rows

#pragma unroll 1
  for (int t = gwave; t < tilesTotal; t += nwav) {
    const int b    = t / (NN / 32);
    const int row0 = (t % (NN / 32)) * 32;

    v8f acc[8];                  // [half][j]: half0 = rows 0..15, half1 = 16..31
#pragma unroll
    for (int j = 0; j < 8; ++j)
#pragma unroll
      for (int r = 0; r < 8; ++r) acc[j][r] = 0.0f;

    // lanes l and l+16 share a source row (same M), cover disjoint K halves
    const long long* arow0 = adjc + (size_t)(row0 + col) * KK;
    const long long* arow1 = adjc + (size_t)(row0 + 16 + col) * KK;
    const float* xbase = x + (size_t)b * NN * FIN;

    // KEEP ROLLED: full unroll makes all loop-invariant B-fragment loads
    // hoistable out of the tile loop -> register spills.
#pragma unroll 1
    for (int k = 0; k < KK; ++k) {
      const int src0 = (int)arow0[k];
      const int src1 = (int)arow1[k];
      const float* xr0 = xbase + (size_t)src0 * FIN;
      const float* xr1 = xbase + (size_t)src1 * FIN;

      // Branch-free prefetch of the next neighbor rows (uniform s_cselect)
      const int kn = (k + 1 < KK) ? (k + 1) : k;
      const float* xn0 = xbase + (size_t)((int)arow0[kn]) * FIN + abase;
      const float* xn1 = xbase + (size_t)((int)arow1[kn]) * FIN + abase;
      __builtin_prefetch(xn0, 0, 3);        // global_prefetch_b8
      __builtin_prefetch(xn0 + 32, 0, 3);
      __builtin_prefetch(xn1, 0, 3);
      __builtin_prefetch(xn1 + 32, 0, 3);

      // Gather rows and pack into four A fragments (2 row-halves x 2 K-slabs).
      // Lane holds K-runs {32q+kb .. 32q+kb+15} (two aligned 32B loads each).
      v8f r00 = *(const v8f*)(xr0 + kb);
      v8f r01 = *(const v8f*)(xr0 + 8 + kb);
      v8f r02 = *(const v8f*)(xr0 + 32 + kb);
      v8f r03 = *(const v8f*)(xr0 + 40 + kb);
      v8f r10 = *(const v8f*)(xr1 + kb);
      v8f r11 = *(const v8f*)(xr1 + 8 + kb);
      v8f r12 = *(const v8f*)(xr1 + 32 + kb);
      v8f r13 = *(const v8f*)(xr1 + 40 + kb);

      v16bf a0, a1, a2, a3;
#pragma unroll
      for (int i = 0; i < 8; ++i) {
        a0[i]     = (__bf16)r00[i];
        a0[8 + i] = (__bf16)r01[i];
        a1[i]     = (__bf16)r02[i];
        a1[8 + i] = (__bf16)r03[i];
        a2[i]     = (__bf16)r10[i];
        a2[8 + i] = (__bf16)r11[i];
        a3[i]     = (__bf16)r12[i];
        a3[8 + i] = (__bf16)r13[i];
      }

      const __bf16* wk = Wt + (k << 12);
#pragma unroll
      for (int j = 0; j < 4; ++j) {
        // B fragment: lane = column (j*16+col), 16 consecutive K -> contiguous 32B
        const __bf16* wc = wk + ((j * 16 + col) << 6);
        v16bf b0 = *(const v16bf*)(wc + kb);        // K = kb .. kb+15
        v16bf b1 = *(const v16bf*)(wc + 32 + kb);   // K = 32+kb .. 32+kb+15
        acc[j] = __builtin_amdgcn_wmma_f32_16x16x32_bf16(
            false, a0, false, b0, (short)0, acc[j], false, false);
        acc[4 + j] = __builtin_amdgcn_wmma_f32_16x16x32_bf16(
            false, a2, false, b0, (short)0, acc[4 + j], false, false);
        acc[j] = __builtin_amdgcn_wmma_f32_16x16x32_bf16(
            false, a1, false, b1, (short)0, acc[j], false, false);
        acc[4 + j] = __builtin_amdgcn_wmma_f32_16x16x32_bf16(
            false, a3, false, b1, (short)0, acc[4 + j], false, false);
      }
    }

    // C/D layout: VGPR r -> M = r + 8*hi (+16 for second row-half), N = col
    float* orow = out + ((size_t)b * NN + (size_t)row0) * FOUT;
#pragma unroll
    for (int h = 0; h < 2; ++h) {
#pragma unroll
      for (int j = 0; j < 4; ++j) {
#pragma unroll
        for (int r = 0; r < 8; ++r) {
          int m = h * 16 + r + hi * 8;
          orow[(size_t)m * FOUT + j * 16 + col] = acc[h * 4 + j][r] + bcol[j];
        }
      }
    }
  }
}

extern "C" void kernel_launch(void* const* d_in, const int* in_sizes, int n_in,
                              void* d_out, int out_size, void* d_ws, size_t ws_size,
                              hipStream_t stream) {
  const float*     x    = (const float*)d_in[0];
  const long long* adjc = (const long long*)d_in[1];
  const float*     W    = (const float*)d_in[2];
  const float*     bias = (const float*)d_in[3];
  float*  out = (float*)d_out;
  __bf16* Wt  = (__bf16*)d_ws;   // 9*64*64*2 = 73728 bytes

  nhconv_prep_w<<<(KK * FIN * FOUT + 255) / 256, 256, 0, stream>>>(W, Wt);

  // 12288 tiles of 32 rows; 384 blocks * 8 waves = 3072 waves -> 4 tiles/wave
  nhconv_main<<<384, 256, 0, stream>>>(x, adjc, Wt, bias, out);
}
